// LNN_47296179864015
// MI455X (gfx1250) — matrix operand order repaired and checked by reading, not compiled, not run on hardware
//
#include <hip/hip_runtime.h>

// ============================================================================
// Lagrangian-NN dynamics on MI455X (gfx1250), forward-over-forward Hessian.
//
// Per sample: z=[q;qdot] in R^4 -> 200 -> 200 -> 200 -> L (softplus MLP).
// g_k = w4.(s3' u^k_a3);  H_jk = w4.(s3'' u^j_a3 u^k_a3 + s3' v^jk_a3)
// with first/second-order forward tangents u, v propagated through W2/W3 in
// forward orientation only (no transposed-weight GEMMs needed).
//
// ~290 GFLOP vs ~4MB I/O -> compute bound -> f16 WMMA, f32 accumulate.
// 2x2 register blocking (2 chains x 2 M-tiles per wave job) cuts LDS traffic
// to 2 ds_load_b128 per v_wmma (A and B fragments each reused twice), matching
// estimated LDS bandwidth per WGP. One 512-thread workgroup per WGP using
// ~322KB LDS (f16 weights + all inter-layer staging), persistent over tiles.
// ============================================================================

typedef __attribute__((ext_vector_type(16))) _Float16 v16h;
typedef __attribute__((ext_vector_type(8)))  _Float16 v8h;
typedef __attribute__((ext_vector_type(8)))  float    v8f;

#define HDIM 200
#define KP   224          // padded K/feature dim (7 * 32)
#define MP   208          // padded output-feature dim (13 * 16)
#define MT   13           // M tiles per GEMM
#define KT   7            // K tiles per GEMM
#define NB   16           // batch elements per tile (WMMA N)
#define THREADS 512
#define NWAVES  (THREADS/32)

union H16 { v16h v; v8h h[2]; };

__device__ __forceinline__ float sigm(float x)  { return 1.0f/(1.0f + __expf(-x)); }
__device__ __forceinline__ float softp(float x) {
  return x > 0.0f ? x + log1pf(__expf(-x)) : log1pf(__expf(x));
}
__device__ __forceinline__ float wave_sum(float v){
  #pragma unroll
  for (int o = 16; o > 0; o >>= 1) v += __shfl_xor(v, o, 32);
  return v;
}

__device__ __forceinline__ v8f wmma32(const H16& a, const H16& b, v8f c){
  return __builtin_amdgcn_wmma_f32_16x16x32_f16(false, a.v, false, b.v,
                                                (short)0, c, false, false);
}

// 2x2 register-blocked GEMM job: chains {A0[,A1]} x M-tiles {m0[,m1]}.
// A: staged activations [batch][KP] f16; B: W rows (row-major LDS) as B-operand.
// DS traffic: TWO&&M2 -> 8 b128 loads per 4 WMMAs (2 per WMMA).
template<bool TWO, bool M2>
__device__ __forceinline__ void gemm_block(const _Float16* A0p, const _Float16* A1p,
                                           const _Float16* W, int m0, int m1, int lane,
                                           v8f& c00, v8f& c01, v8f& c10, v8f& c11)
{
  const int r  = lane & 15;        // A batch row / B column-in-tile
  const int hi = lane >> 4;
  const _Float16* a0 = A0p + r*KP + hi*8;           // A: K chunks hi*8 & 16+hi*8
  const _Float16* a1 = A1p + r*KP + hi*8;
  const _Float16* b0 = W + (m0*16 + r)*KP + hi*8;   // B: K chunks hi*8 & 16+hi*8
  const _Float16* b1 = W + (m1*16 + r)*KP + hi*8;
  #pragma unroll
  for (int kt = 0; kt < KT; ++kt){
    H16 Af, Bf;
    Af.h[0] = *(const v8h*)(a0 + kt*32);
    Af.h[1] = *(const v8h*)(a0 + kt*32 + 16);
    Bf.h[0] = *(const v8h*)(b0 + kt*32);
    Bf.h[1] = *(const v8h*)(b0 + kt*32 + 16);
    c00 = wmma32(Af, Bf, c00);
    H16 B2f;
    if (M2){
      B2f.h[0] = *(const v8h*)(b1 + kt*32);
      B2f.h[1] = *(const v8h*)(b1 + kt*32 + 16);
      c01 = wmma32(Af, B2f, c01);
    }
    if (TWO){
      H16 A2f;
      A2f.h[0] = *(const v8h*)(a1 + kt*32);
      A2f.h[1] = *(const v8h*)(a1 + kt*32 + 16);
      c10 = wmma32(A2f, Bf, c10);
      if (M2) c11 = wmma32(A2f, B2f, c11);
    }
  }
}

// C fragment: lane holds feature n = m*16+(lane&15), c[j] = batch (lane>>4)*8+j.
__device__ __forceinline__ void store_tile(_Float16* dst, v8f c, int m, int lane,
                                           const float* bias){
  const int n  = m*16 + (lane & 15);
  const int b0 = (lane >> 4) * 8;
  const float bv = bias ? bias[n] : 0.0f;
  #pragma unroll
  for (int j = 0; j < 8; ++j)
    dst[(b0 + j)*KP + n] = (_Float16)(c[j] + bv);
}

// One GEMM level: C chains, src = stg[srcB+c], dst = stg[dstB+c], bias on chain 0.
// Jobs = chain-pairs x 7 m-pairs, distributed over waves (wave-uniform branches).
__device__ __forceinline__ void run_level(_Float16 (*stg)[NB*KP], int C,
                                          int srcB, int dstB,
                                          const _Float16* W, const float* bias0,
                                          int wave, int lane)
{
  const int ncp = (C + 1) >> 1;
  for (int j = wave; j < ncp*7; j += NWAVES){
    const int cp = j / 7, mp = j - cp*7;
    const int c0 = cp*2;
    const int m0 = mp*2;
    const int m1 = (mp == 6) ? 12 : m0 + 1;
    const bool two = (c0 + 1) < C;
    const bool m2  = (m1 != m0);
    v8f c00 = {0.f,0.f,0.f,0.f,0.f,0.f,0.f,0.f};
    v8f c01 = c00, c10 = c00, c11 = c00;
    const _Float16* A0 = &stg[srcB + c0][0];
    const _Float16* A1 = &stg[srcB + (two ? c0 + 1 : c0)][0];
    if (two){ if (m2) gemm_block<true , true >(A0, A1, W, m0, m1, lane, c00, c01, c10, c11);
              else    gemm_block<true , false>(A0, A1, W, m0, m1, lane, c00, c01, c10, c11); }
    else    { if (m2) gemm_block<false, true >(A0, A1, W, m0, m1, lane, c00, c01, c10, c11);
              else    gemm_block<false, false>(A0, A1, W, m0, m1, lane, c00, c01, c10, c11); }
    const float* bs0 = (c0 == 0) ? bias0 : (const float*)nullptr;
    store_tile(&stg[dstB + c0][0], c00, m0, lane, bs0);
    if (m2) store_tile(&stg[dstB + c0][0], c01, m1, lane, bs0);
    if (two){
      store_tile(&stg[dstB + c0 + 1][0], c10, m0, lane, nullptr);
      if (m2) store_tile(&stg[dstB + c0 + 1][0], c11, m1, lane, nullptr);
    }
  }
}

__global__ __launch_bounds__(THREADS, 1)
void LNN_47296179864015_kernel(const float* __restrict__ x,
                               const float* __restrict__ W1g, const float* __restrict__ b1g,
                               const float* __restrict__ W2g, const float* __restrict__ b2g,
                               const float* __restrict__ W3g, const float* __restrict__ b3g,
                               const float* __restrict__ W4g,
                               float* __restrict__ out, int nTiles)
{
  // ---- LDS: ~321.6 KB total (one workgroup per WGP) ----
  __shared__ alignas(16) _Float16 W2L[MP*KP];      // 93,184 B
  __shared__ alignas(16) _Float16 W3L[MP*KP];      // 93,184 B
  // STG[0..4]  persist: a2, u_a2[0..3]
  // STG[5..9]  persist: a3, u_a3[0..3]
  // STG[10..14] region A (GEMM inputs: h-level / v_h-level)
  // STG[15..17] region B (v_a2 / v_a3 outputs)
  __shared__ alignas(16) _Float16 STG[18][NB*KP];  // 129,024 B
  __shared__ alignas(16) _Float16 W1c[4*KP];       // 1,792 B (W1 columns)
  __shared__ float b1L[KP], b2L[KP], b3L[KP], w4L[KP];  // 3,584 B
  __shared__ float xq[NB*4];                        // 256 B
  __shared__ float Hacc[NB][9];                     // 576 B

  const int tid  = threadIdx.x;
  const int lane = tid & 31;
  const int wave = tid >> 5;

  // ---- one-time: weights -> LDS f16, zero-padded (padding garbage-safe) ----
  for (int i = tid; i < MP*KP; i += THREADS){
    int r = i / KP, k = i - r*KP;
    bool v = (r < HDIM) && (k < HDIM);
    W2L[i] = v ? (_Float16)W2g[r*HDIM + k] : (_Float16)0.0f;
    W3L[i] = v ? (_Float16)W3g[r*HDIM + k] : (_Float16)0.0f;
  }
  for (int f = tid; f < KP; f += THREADS){
    bool v = f < HDIM;
    W1c[0*KP+f] = v ? (_Float16)W1g[f*4+0] : (_Float16)0.0f;
    W1c[1*KP+f] = v ? (_Float16)W1g[f*4+1] : (_Float16)0.0f;
    W1c[2*KP+f] = v ? (_Float16)W1g[f*4+2] : (_Float16)0.0f;
    W1c[3*KP+f] = v ? (_Float16)W1g[f*4+3] : (_Float16)0.0f;
    b1L[f] = v ? b1g[f] : 0.0f;
    b2L[f] = v ? b2g[f] : 0.0f;
    b3L[f] = v ? b3g[f] : 0.0f;
    w4L[f] = v ? W4g[f] : 0.0f;
  }
  // zero staging once so K-tails [208,224) never feed NaN garbage into WMMA
  for (int i = tid; i < 18*NB*KP; i += THREADS) (&STG[0][0])[i] = (_Float16)0.0f;
  __syncthreads();

  // second-order pair list: (j,k) -> Hacc slot (index in this list)
  constexpr int pj[7] = {0,1,2,0,1,2,3};
  constexpr int pk[7] = {2,2,2,3,3,3,3};

  for (int tile = blockIdx.x; tile < nTiles; tile += gridDim.x){
    if (tid < NB*4) xq[tid] = x[(size_t)tile*(NB*4) + tid];
    if (tile + (int)gridDim.x < nTiles && tid < NB*4)   // speculative prefetch
      __builtin_prefetch(&x[(size_t)(tile + gridDim.x)*(NB*4) + tid], 0, 1);
    __syncthreads();

    // ================= pass 0: primal + 4 first-order tangents =================
    for (int i = tid; i < NB*MP; i += THREADS){
      int b = i / MP, f = i - b*MP;
      const float* xb = &xq[b*4];
      float a1 = b1L[f] + (float)W1c[f]*xb[0] + (float)W1c[KP+f]*xb[1]
               + (float)W1c[2*KP+f]*xb[2] + (float)W1c[3*KP+f]*xb[3];
      float sg = sigm(a1);
      int o = b*KP + f;
      STG[10][o] = (_Float16)softp(a1);                    // h1
      #pragma unroll
      for (int k = 0; k < 4; ++k)
        STG[11+k][o] = (_Float16)(sg * (float)W1c[k*KP+f]); // u^k_h1 = s1' W1[:,k]
    }
    __syncthreads();
    run_level(STG, 5, 10, 0, W2L, b2L, wave, lane);         // a2 / u_a2
    __syncthreads();
    for (int i = tid; i < NB*MP; i += THREADS){             // elementwise @ level 2
      int b = i / MP, f = i - b*MP;
      int o = b*KP + f;
      float a2 = (float)STG[0][o];
      float sg = sigm(a2);
      STG[10][o] = (_Float16)softp(a2);                     // h2
      #pragma unroll
      for (int k = 0; k < 4; ++k)
        STG[11+k][o] = (_Float16)(sg * (float)STG[1+k][o]); // u^k_h2
    }
    __syncthreads();
    run_level(STG, 5, 10, 5, W3L, b3L, wave, lane);         // a3 / u_a3
    __syncthreads();
    {                                                       // g0,g1 = w4.(s3' u^k_a3)
      int b = wave;
      float g0 = 0.f, g1 = 0.f;
      for (int f = lane; f < HDIM; f += 32){
        int o = b*KP + f;
        float wf = w4L[f] * sigm((float)STG[5][o]);
        g0 += wf * (float)STG[6][o];
        g1 += wf * (float)STG[7][o];
      }
      g0 = wave_sum(g0); g1 = wave_sum(g1);
      if (lane == 0){ Hacc[b][7] = g0; Hacc[b][8] = g1; }
    }
    __syncthreads();

    // ============== passes 1..3: second-order pairs (3 + 3 + 1) ==============
    #pragma unroll
    for (int pp = 0; pp < 3; ++pp){
      const int c0 = pp*3;
      const int nc = (pp < 2) ? 3 : 1;
      for (int i = tid; i < NB*MP; i += THREADS){           // v_h1 = s1'' cj ck
        int b = i / MP, f = i - b*MP;
        const float* xb = &xq[b*4];
        float a1 = b1L[f] + (float)W1c[f]*xb[0] + (float)W1c[KP+f]*xb[1]
                 + (float)W1c[2*KP+f]*xb[2] + (float)W1c[3*KP+f]*xb[3];
        float sg = sigm(a1);
        float sd = sg * (1.0f - sg);
        int o = b*KP + f;
        #pragma unroll
        for (int c = 0; c < nc; ++c){
          float cj = (float)W1c[pj[c0+c]*KP + f];
          float ck = (float)W1c[pk[c0+c]*KP + f];
          STG[10+c][o] = (_Float16)(sd * cj * ck);
        }
      }
      __syncthreads();
      run_level(STG, nc, 10, 15, W2L, nullptr, wave, lane); // v_a2 = W2 v_h1
      __syncthreads();
      for (int i = tid; i < NB*MP; i += THREADS){           // v_h2
        int b = i / MP, f = i - b*MP;
        int o = b*KP + f;
        float a2 = (float)STG[0][o];
        float sg = sigm(a2);
        float sd = sg * (1.0f - sg);
        #pragma unroll
        for (int c = 0; c < nc; ++c){
          float uj = (float)STG[1 + pj[c0+c]][o];
          float uk = (float)STG[1 + pk[c0+c]][o];
          STG[10+c][o] = (_Float16)(sd*uj*uk + sg*(float)STG[15+c][o]);
        }
      }
      __syncthreads();
      run_level(STG, nc, 10, 15, W3L, nullptr, wave, lane); // v_a3 = W3 v_h2
      __syncthreads();
      {                                                     // H_jk reductions
        int b = wave;
        float h0 = 0.f, h1 = 0.f, h2 = 0.f;
        for (int f = lane; f < HDIM; f += 32){
          int o = b*KP + f;
          float a3 = (float)STG[5][o];
          float sg = sigm(a3);
          float sd = sg * (1.0f - sg);
          float wf = w4L[f];
          {
            float uj = (float)STG[6+pj[c0]][o], uk = (float)STG[6+pk[c0]][o];
            h0 += wf*(sd*uj*uk + sg*(float)STG[15][o]);
          }
          if (nc > 1){
            float uj = (float)STG[6+pj[c0+1]][o], uk = (float)STG[6+pk[c0+1]][o];
            h1 += wf*(sd*uj*uk + sg*(float)STG[16][o]);
            uj = (float)STG[6+pj[c0+2]][o]; uk = (float)STG[6+pk[c0+2]][o];
            h2 += wf*(sd*uj*uk + sg*(float)STG[17][o]);
          }
        }
        h0 = wave_sum(h0);
        if (lane == 0) Hacc[b][c0] = h0;
        if (nc > 1){
          h1 = wave_sum(h1); h2 = wave_sum(h2);
          if (lane == 0){ Hacc[b][c0+1] = h1; Hacc[b][c0+2] = h2; }
        }
      }
      __syncthreads();
    }

    // ==================== 2x2 regularized solve + store ====================
    if (tid < NB){
      const int b = tid;
      float qd0 = xq[b*4+2], qd1 = xq[b*4+3];
      float H02 = Hacc[b][0], H12 = Hacc[b][1], H22 = Hacc[b][2];
      float H03 = Hacc[b][3], H13 = Hacc[b][4], H23 = Hacc[b][5], H33 = Hacc[b][6];
      float g0  = Hacc[b][7], g1  = Hacc[b][8];
      float r0 = g0 - (H02*qd0 + H12*qd1);     // rhs = dL/dq - mix @ qdot
      float r1 = g1 - (H03*qd0 + H13*qd1);
      float M00 = H22 + 1e-6f, M01 = H23, M11 = H33 + 1e-6f;
      float inv = 1.0f / (M00*M11 - M01*M01);
      float4 o4;
      o4.x = qd0; o4.y = qd1;
      o4.z = (M11*r0 - M01*r1) * inv;
      o4.w = (M00*r1 - M01*r0) * inv;
      *(float4*)(out + (size_t)(tile*NB + b)*4) = o4;
    }
    __syncthreads();   // protect xq/Hacc/STG reuse next tile
  }
}

extern "C" void kernel_launch(void* const* d_in, const int* in_sizes, int n_in,
                              void* d_out, int out_size, void* d_ws, size_t ws_size,
                              hipStream_t stream) {
  (void)n_in; (void)out_size; (void)d_ws; (void)ws_size;
  const float* x  = (const float*)d_in[0];
  const float* W1 = (const float*)d_in[1];
  const float* b1 = (const float*)d_in[2];
  const float* W2 = (const float*)d_in[3];
  const float* b2 = (const float*)d_in[4];
  const float* W3 = (const float*)d_in[5];
  const float* b3 = (const float*)d_in[6];
  const float* W4 = (const float*)d_in[7];
  float* out = (float*)d_out;

  const int Bsz    = in_sizes[0] / 4;      // (B, 2D) with D=2
  const int nTiles = Bsz / NB;             // B divisible by 16
  if (nTiles <= 0) return;
  int grid = nTiles < 512 ? nTiles : 512;  // persistent workgroups (1 per WGP)
  LNN_47296179864015_kernel<<<dim3(grid), dim3(THREADS), 0, stream>>>(
      x, W1, b1, W2, b2, W3, b3, W4, out, nTiles);
}
